// BasicGNN_5677946765815
// MI455X (gfx1250) — matrix operand order repaired and checked by reading, compiled, not run on hardware
//
#include <hip/hip_runtime.h>
#include <hip/hip_bf16.h>

typedef __attribute__((ext_vector_type(16))) __bf16 v16bf;
typedef __attribute__((ext_vector_type(8)))  __bf16 v8bf;
typedef __attribute__((ext_vector_type(8)))  float  v8f;

#define WT_STRIDE 72   // padded bf16 row stride for transposed W in LDS

// Concatenate two 8-wide bf16 halves into a 16-wide fragment (no moves).
__device__ __forceinline__ v16bf cat8(v8bf lo, v8bf hi) {
    return __builtin_shufflevector(lo, hi, 0, 1, 2, 3, 4, 5, 6, 7,
                                           8, 9, 10, 11, 12, 13, 14, 15);
}

__device__ __forceinline__ void pack8(v16bf& a, int base, float4 p, float4 q) {
    a[base + 0] = (__bf16)p.x; a[base + 1] = (__bf16)p.y;
    a[base + 2] = (__bf16)p.z; a[base + 3] = (__bf16)p.w;
    a[base + 4] = (__bf16)q.x; a[base + 5] = (__bf16)q.y;
    a[base + 6] = (__bf16)q.z; a[base + 7] = (__bf16)q.w;
}

// ---------------------------------------------------------------- utilities
__global__ void zero4_kernel(float4* __restrict__ p, size_t n4) {
    size_t i = (size_t)blockIdx.x * blockDim.x + threadIdx.x;
    if (i < n4) p[i] = make_float4(0.f, 0.f, 0.f, 0.f);
}

__global__ void init_deg_kernel(float* __restrict__ deg, int n) {
    int i = blockIdx.x * blockDim.x + threadIdx.x;
    if (i < n) deg[i] = 1.0f;                      // self-loop weight
}

__global__ void deg_acc_kernel(const int* __restrict__ dst,
                               const float* __restrict__ ew,
                               float* __restrict__ deg, int nE) {
    int e = blockIdx.x * blockDim.x + threadIdx.x;
    if (e < nE) atomicAdd(&deg[dst[e]], ew[e]);
}

__global__ void make_dis_kernel(float* __restrict__ deg, int n) {
    int i = blockIdx.x * blockDim.x + threadIdx.x;
    if (i < n) {
        float d = deg[i];
        deg[i] = (d > 0.f) ? rsqrtf(d) : 0.f;      // in place: deg -> dis
    }
}

__global__ void make_norm_kernel(const int* __restrict__ src,
                                 const int* __restrict__ dst,
                                 const float* __restrict__ ew,
                                 const float* __restrict__ dis,
                                 float* __restrict__ nrm, int nE) {
    int e = blockIdx.x * blockDim.x + threadIdx.x;
    if (e < nE) nrm[e] = dis[src[e]] * dis[dst[e]] * ew[e];
}

// -------------------------------------------------- node GEMM: H = cur @ W
// 128 threads = 4 waves, each wave computes a 16-row x 64-col strip of H.
__global__ __launch_bounds__(128) void gcn_gemm_kernel(
        const float* __restrict__ X, const float* __restrict__ W,
        float* __restrict__ H, int nRows) {
    __shared__ __bf16 Wt[64 * WT_STRIDE];          // transposed: Wt[n][k]
    int t = threadIdx.x;
    for (int i = t; i < 64 * 64; i += 128) {
        int k = i >> 6, n = i & 63;
        Wt[n * WT_STRIDE + k] = (__bf16)W[i];
    }
    __syncthreads();

    int wave = t >> 5;
    int lane = t & 31;
    int strip = blockIdx.x * 4 + wave;             // wave-uniform
    if (strip * 16 >= nRows) return;               // uniform exit, WMMA-safe

    int mrow = strip * 16 + (lane & 15);
    int mrowc = mrow < nRows ? mrow : (nRows - 1); // clamp loads (full-EXEC)
    int ncol = lane & 15;

    // B fragments: lane = column n; lanes<16 hold K base+0..7 & base+16..23; lanes>=16 +8.
    // Transposed+padded LDS -> two aligned 16B loads per fragment.
    v16bf Bf[4][2];
    #pragma unroll
    for (int nt = 0; nt < 4; ++nt)
        #pragma unroll
        for (int ks = 0; ks < 2; ++ks) {
            int kbase = ks * 32 + ((lane >> 4) << 3);
            const __bf16* wp = &Wt[(nt * 16 + ncol) * WT_STRIDE + kbase];
            Bf[nt][ks] = cat8(*(const v8bf*)wp, *(const v8bf*)(wp + 16));
        }

    v8f acc[4] = {};
    const float* xr = X + (size_t)mrowc * 64;
    #pragma unroll
    for (int ks = 0; ks < 2; ++ks) {
        // A layout: lanes<16 hold K base+0..7 & base+16..23; lanes>=16 +8.
        int kb = ks * 32 + ((lane >> 4) << 3);
        float4 f0 = *(const float4*)(xr + kb);
        float4 f1 = *(const float4*)(xr + kb + 4);
        float4 f2 = *(const float4*)(xr + kb + 16);
        float4 f3 = *(const float4*)(xr + kb + 20);
        v16bf a;
        pack8(a, 0, f0, f1);
        pack8(a, 8, f2, f3);
        #pragma unroll
        for (int nt = 0; nt < 4; ++nt)
            acc[nt] = __builtin_amdgcn_wmma_f32_16x16x32_bf16(
                false, a, false, Bf[nt][ks], (short)0, acc[nt], false, false);
    }

    // C/D layout: VGPR i -> row i (lanes<16) / row 8+i (lanes>=16), col=lane%16
    int rowOff = (lane >> 4) * 8;
    #pragma unroll
    for (int nt = 0; nt < 4; ++nt)
        #pragma unroll
        for (int i = 0; i < 8; ++i) {
            int r = strip * 16 + rowOff + i;
            if (r < nRows) H[(size_t)r * 64 + nt * 16 + ncol] = acc[nt][i];
        }
}

// ---------------------------------------- edge scatter: agg[dst] += h[src]*w
// 256 threads = 8 waves; one wave per edge, each lane moves a float2.
__global__ __launch_bounds__(256) void gcn_scatter_kernel(
        const int* __restrict__ src, const int* __restrict__ dst,
        const float* __restrict__ nrm, const float* __restrict__ H,
        float* __restrict__ AGG, int nE) {
    int wave = threadIdx.x >> 5;
    int lane = threadIdx.x & 31;
    int e = blockIdx.x * 8 + wave;
    if (e >= nE) return;
    int s = src[e], d = dst[e];
    float w = nrm[e];
    float2 v = ((const float2*)(H + (size_t)s * 64))[lane];
    float* ap = AGG + (size_t)d * 64 + lane * 2;
    atomicAdd(ap,     v.x * w);
    atomicAdd(ap + 1, v.y * w);
}

// ------------------------- epilogue: out = act(agg + h*dis^2 (self-loop) + b)
__global__ void bias_act_kernel(const float* __restrict__ AGG,
                                const float* __restrict__ H,
                                const float* __restrict__ dis,
                                const float* __restrict__ bias,
                                float* __restrict__ OUT,
                                size_t total, int doTanh) {
    size_t i = (size_t)blockIdx.x * blockDim.x + threadIdx.x;
    if (i >= total) return;
    size_t node = i >> 6;
    int c = (int)(i & 63);
    float dn = dis[node];
    float v = AGG[i] + H[i] * dn * dn + bias[c];
    OUT[i] = doTanh ? tanhf(v) : v;
}

// ----------------------------------------------------- mean-pool accumulation
__global__ __launch_bounds__(256) void pool_acc_kernel(
        const float* __restrict__ H, const int* __restrict__ batch,
        float* __restrict__ psum, float* __restrict__ cnt, int n) {
    int wave = threadIdx.x >> 5;
    int lane = threadIdx.x & 31;
    int node = blockIdx.x * 8 + wave;
    if (node >= n) return;
    int g = batch[node];
    float2 v = ((const float2*)(H + (size_t)node * 64))[lane];
    atomicAdd(&psum[g * 64 + lane * 2],     v.x);
    atomicAdd(&psum[g * 64 + lane * 2 + 1], v.y);
    if (lane == 0) atomicAdd(&cnt[g], 1.0f);
}

// ---------------------------------------------------------------- FC head
struct FCParams { const float* W[6]; const float* B[6]; };

__global__ __launch_bounds__(128) void fc_head_kernel(
        const float* __restrict__ psum, const float* __restrict__ cnt,
        FCParams p, float* __restrict__ out) {
    __shared__ float  act[64 * 64];
    __shared__ __bf16 Wt[64 * WT_STRIDE];          // transposed: Wt[n][k]
    int t = threadIdx.x, wave = t >> 5, lane = t & 31;
    int ncol = lane & 15;

    for (int i = t; i < 4096; i += 128) {
        int g = i >> 6;
        act[i] = psum[i] / fmaxf(cnt[g], 1.0f);
    }
    __syncthreads();

    for (int l = 0; l < 6; ++l) {
        const float* Wg = p.W[l];
        for (int i = t; i < 4096; i += 128) {
            int k = i >> 6, n = i & 63;
            Wt[n * WT_STRIDE + k] = (__bf16)Wg[i];
        }
        __syncthreads();

        v16bf Bf[4][2];
        #pragma unroll
        for (int nt = 0; nt < 4; ++nt)
            #pragma unroll
            for (int ks = 0; ks < 2; ++ks) {
                int kbase = ks * 32 + ((lane >> 4) << 3);
                const __bf16* wp = &Wt[(nt * 16 + ncol) * WT_STRIDE + kbase];
                Bf[nt][ks] = cat8(*(const v8bf*)wp, *(const v8bf*)(wp + 16));
            }

        v8f acc[4] = {};
        int row = wave * 16 + (lane & 15);
        const float* ar = &act[row * 64];
        #pragma unroll
        for (int ks = 0; ks < 2; ++ks) {
            int kb = ks * 32 + ((lane >> 4) << 3);
            float4 f0 = *(const float4*)(ar + kb);
            float4 f1 = *(const float4*)(ar + kb + 4);
            float4 f2 = *(const float4*)(ar + kb + 16);
            float4 f3 = *(const float4*)(ar + kb + 20);
            v16bf a;
            pack8(a, 0, f0, f1);
            pack8(a, 8, f2, f3);
            #pragma unroll
            for (int nt = 0; nt < 4; ++nt)
                acc[nt] = __builtin_amdgcn_wmma_f32_16x16x32_bf16(
                    false, a, false, Bf[nt][ks], (short)0, acc[nt], false, false);
        }
        __syncthreads();  // done reading act

        const float* bias = p.B[l];
        int rowOff = (lane >> 4) * 8;
        #pragma unroll
        for (int nt = 0; nt < 4; ++nt) {
            float bv = bias[nt * 16 + ncol];
            #pragma unroll
            for (int i = 0; i < 8; ++i) {
                int r = wave * 16 + rowOff + i;
                act[r * 64 + nt * 16 + ncol] = fmaxf(acc[nt][i] + bv, 0.0f);
            }
        }
        __syncthreads();
    }
    for (int i = t; i < 4096; i += 128) out[i] = act[i];
}

// ---------------------------------------------------------------- launcher
extern "C" void kernel_launch(void* const* d_in, const int* in_sizes, int n_in,
                              void* d_out, int out_size, void* d_ws, size_t ws_size,
                              hipStream_t stream) {
    (void)n_in; (void)out_size; (void)ws_size;
    const float* x     = (const float*)d_in[0];
    const int*   ei    = (const int*)d_in[1];
    const float* ea    = (const float*)d_in[2];
    const int*   batch = (const int*)d_in[3];

    int N = in_sizes[0] / 64;
    int E = in_sizes[1] / 2;
    const int* src = ei;
    const int* dst = ei + E;

    size_t N64 = (size_t)N * 64;
    float* ws   = (float*)d_ws;
    float* bufH = ws;               // GEMM result h           [N*64]
    float* bufA = bufH + N64;       // ping                    [N*64]
    float* bufB = bufA + N64;       // pong                    [N*64]
    float* dis  = bufB + N64;       // deg -> dis              [N]
    float* nrm  = dis + N;          // per-edge norm           [E]
    float* psum = nrm + E;          // pooled sums             [64*64]
    float* cnt  = psum + 64 * 64;   // pooled counts           [64]

    const int TB = 256;

    // ---- degree / normalization precompute (once per call) ----
    init_deg_kernel<<<(N + TB - 1) / TB, TB, 0, stream>>>(dis, N);
    deg_acc_kernel<<<(E + TB - 1) / TB, TB, 0, stream>>>(dst, ea, dis, E);
    make_dis_kernel<<<(N + TB - 1) / TB, TB, 0, stream>>>(dis, N);
    make_norm_kernel<<<(E + TB - 1) / TB, TB, 0, stream>>>(src, dst, ea, dis, nrm, E);

    // ---- 6 GCN layers ----
    const float* cur = x;
    int strips = (N + 15) / 16;
    int gemmBlocks = (strips + 3) / 4;
    size_t n4 = N64 / 4;
    for (int l = 0; l < 6; ++l) {
        const float* W = (const float*)d_in[4 + 2 * l];
        const float* b = (const float*)d_in[5 + 2 * l];
        float* agg = (l & 1) ? bufB : bufA;

        zero4_kernel<<<(unsigned)((n4 + TB - 1) / TB), TB, 0, stream>>>((float4*)agg, n4);
        gcn_gemm_kernel<<<gemmBlocks, 128, 0, stream>>>(cur, W, bufH, N);
        gcn_scatter_kernel<<<(E + 7) / 8, 256, 0, stream>>>(src, dst, nrm, bufH, agg, E);
        bias_act_kernel<<<(unsigned)((N64 + TB - 1) / TB), TB, 0, stream>>>(
            agg, bufH, dis, b, agg, N64, (l != 2) ? 1 : 0);
        cur = agg;
    }

    // ---- global mean pool ----
    size_t pz4 = (64 * 64 + 64) / 4;
    zero4_kernel<<<(unsigned)((pz4 + TB - 1) / TB), TB, 0, stream>>>((float4*)psum, pz4);
    pool_acc_kernel<<<(N + 7) / 8, 256, 0, stream>>>(cur, batch, psum, cnt, N);

    // ---- FC head ----
    FCParams fp;
    for (int l = 0; l < 6; ++l) {
        fp.W[l] = (const float*)d_in[16 + 2 * l];
        fp.B[l] = (const float*)d_in[17 + 2 * l];
    }
    fc_head_kernel<<<1, 128, 0, stream>>>(psum, cnt, fp, (float*)d_out);
}
